// LinearQuant_54030688583739
// MI455X (gfx1250) — compile-verified, matrix operand order checked by reading, not run on hardware
//
#include <hip/hip_runtime.h>
#include <stdint.h>

typedef __attribute__((ext_vector_type(16))) _Float16 v16h;
typedef __attribute__((ext_vector_type(8)))  _Float16 v8h;
typedef __attribute__((ext_vector_type(2)))  _Float16 h2;
typedef __attribute__((ext_vector_type(8)))  float    v8f;
typedef __attribute__((ext_vector_type(4)))  float    v4f;
typedef __attribute__((ext_vector_type(4)))  int      v4i;
typedef __attribute__((ext_vector_type(4)))  unsigned int v4u;
typedef __attribute__((ext_vector_type(2)))  unsigned int v2u;

#define IN_F  4096
#define OUT_F 11008
#define KSTEP 32
#define LDA   40   // halves per A-tile row (32 + 8 pad -> 80B)
#define LDB   40   // halves per B-tile row

#if __has_builtin(__builtin_amdgcn_global_load_async_to_lds_b128)
#define HAVE_ASYNC 1
#endif

// ---------------- int8 -> f16 dequant via fp16 magic numbers ----------------
// h = bitcast(0x64 : (b^0x80)) = 1024 + (b+128); h - 1152 == (float16)b exactly.
__device__ __forceinline__ v8h dequant8(unsigned lo, unsigned hi) {
  lo ^= 0x80808080u; hi ^= 0x80808080u;
  const h2 c1152 = { (_Float16)1152.0f, (_Float16)1152.0f };
  v8h r;
#pragma unroll
  for (int p = 0; p < 4; ++p) {
    unsigned src = (p < 2) ? lo : hi;
    unsigned sel = (p & 1) ? 0x07030602u : 0x05010400u;   // {0x64,b(2i+1),0x64,b(2i)}
    unsigned d   = __builtin_amdgcn_perm(0x64646464u, src, sel);
    h2 v = __builtin_bit_cast(h2, d) - c1152;
    r[2 * p] = v[0]; r[2 * p + 1] = v[1];
  }
  return r;
}

// ---------------- prepass kernels ----------------
__global__ __launch_bounds__(256)
void cvt_w_kernel(const int8_t* __restrict__ w, _Float16* __restrict__ dst, int n8) {
  int i = blockIdx.x * 256 + threadIdx.x;
  if (i >= n8) return;
  v2u q = ((const v2u*)w)[i];
  ((v8h*)dst)[i] = dequant8(q[0], q[1]);
}

__global__ __launch_bounds__(256)
void cvt_x_kernel(const float* __restrict__ x, _Float16* __restrict__ dst, int n8) {
  int i = blockIdx.x * 256 + threadIdx.x;
  if (i >= n8) return;
  v4f a = ((const v4f*)x)[2 * i], b = ((const v4f*)x)[2 * i + 1];
  v8h r;
#pragma unroll
  for (int e = 0; e < 4; ++e) { r[e] = (_Float16)a[e]; r[e + 4] = (_Float16)b[e]; }
  ((v8h*)dst)[i] = r;
}

// ---------------- async copy helpers ----------------
#ifdef HAVE_ASYNC
__device__ __forceinline__ void async_cp16(const void* g, void* l) {
  __builtin_amdgcn_global_load_async_to_lds_b128(
      (__attribute__((address_space(1))) v4i*)g,
      (__attribute__((address_space(3))) v4i*)l, 0, 0);
}
__device__ __forceinline__ void wait_async0() {
#if __has_builtin(__builtin_amdgcn_s_wait_asynccnt)
  __builtin_amdgcn_s_wait_asynccnt(0);
#else
  asm volatile("s_wait_asynccnt 0" ::: "memory");
#endif
}
#endif

__device__ __forceinline__ v16h frag_from_lds(const _Float16* p) {
  v8h lo = *(const v8h*)(p);
  v8h hi = *(const v8h*)(p + 16);
  v16h f;
#pragma unroll
  for (int e = 0; e < 8; ++e) { f[e] = lo[e]; f[e + 8] = hi[e]; }
  return f;
}

// ---------------- fast path: both operands already f16 in global ----------------
// Block tile 128x256, 8 waves (2x4), wave tile 64x64 = 4x4 WMMA 16x16x32 f16.
__global__ __launch_bounds__(256, 1)
void lq_wmma_f16_kernel(const _Float16* __restrict__ xa,
                        const _Float16* __restrict__ wb,
                        const float* __restrict__ scaler,
                        float* __restrict__ out)
{
  __shared__ _Float16 smA[2][128 * LDA];
  __shared__ _Float16 smB[2][256 * LDB];

  const int t    = threadIdx.x;
  const int lane = t & 31;
  const int wave = t >> 5;
  const int lrow = lane & 15;
  const int lhi  = lane >> 4;

  const int m0b = blockIdx.y * 128;
  const int n0b = blockIdx.x * 256;
  const int waveM = (wave >> 2) * 64;
  const int waveN = (wave & 3) * 64;

  const int rowA  = t >> 1;   // 16 halves (2x b128) per thread
  const int halfA = t & 1;
  const int rowB  = t;        // 32 halves (4x b128) per thread
  const _Float16* gA = xa + (size_t)(m0b + rowA) * IN_F + halfA * 16;
  const _Float16* gB = wb + (size_t)(n0b + rowB) * IN_F;
  const int baseA = rowA * LDA + halfA * 16;
  const int baseB = rowB * LDB;

#ifdef HAVE_ASYNC
  auto stage_async = [&](int buf, int k) {
    async_cp16(gA + k,     &smA[buf][baseA]);
    async_cp16(gA + k + 8, &smA[buf][baseA + 8]);
#pragma unroll
    for (int c = 0; c < 4; ++c)
      async_cp16(gB + k + 8 * c, &smB[buf][baseB + 8 * c]);
  };
#endif

  v8f acc[4][4];
  const v8f vzero = {0.f, 0.f, 0.f, 0.f, 0.f, 0.f, 0.f, 0.f};
#pragma unroll
  for (int i = 0; i < 4; ++i)
#pragma unroll
    for (int j = 0; j < 4; ++j) acc[i][j] = vzero;

  // prologue: stage k=0 into buffer 0
#ifdef HAVE_ASYNC
  stage_async(0, 0);
  wait_async0();
#else
  {
    v8h a0 = *(const v8h*)(gA), a1 = *(const v8h*)(gA + 8);
    v8h b0 = *(const v8h*)(gB),      b1 = *(const v8h*)(gB + 8);
    v8h b2 = *(const v8h*)(gB + 16), b3 = *(const v8h*)(gB + 24);
    *(v8h*)&smA[0][baseA] = a0; *(v8h*)&smA[0][baseA + 8] = a1;
    *(v8h*)&smB[0][baseB]      = b0; *(v8h*)&smB[0][baseB + 8]  = b1;
    *(v8h*)&smB[0][baseB + 16] = b2; *(v8h*)&smB[0][baseB + 24] = b3;
  }
#endif
  __syncthreads();

  int cur = 0;
  for (int k = 0; k < IN_F; k += KSTEP) {
    const bool has_next = (k + KSTEP < IN_F);

#ifdef HAVE_ASYNC
    if (has_next) stage_async(cur ^ 1, k + KSTEP);   // DMA in flight under WMMA
#else
    v8h a0, a1, b0, b1, b2, b3;
    if (has_next) {
      const _Float16* pA = gA + k + KSTEP;
      const _Float16* pB = gB + k + KSTEP;
      a0 = *(const v8h*)(pA); a1 = *(const v8h*)(pA + 8);
      b0 = *(const v8h*)(pB);      b1 = *(const v8h*)(pB + 8);
      b2 = *(const v8h*)(pB + 16); b3 = *(const v8h*)(pB + 24);
    }
#endif

    v16h a[4], b[4];
#pragma unroll
    for (int i = 0; i < 4; ++i)
      a[i] = frag_from_lds(&smA[cur][(waveM + i * 16 + lrow) * LDA + lhi * 8]);
#pragma unroll
    for (int j = 0; j < 4; ++j)
      b[j] = frag_from_lds(&smB[cur][(waveN + j * 16 + lrow) * LDB + lhi * 8]);

#pragma unroll
    for (int i = 0; i < 4; ++i)
#pragma unroll
      for (int j = 0; j < 4; ++j)
        acc[i][j] = __builtin_amdgcn_wmma_f32_16x16x32_f16(
            false, a[i], false, b[j], (short)0, acc[i][j], false, false);

#ifdef HAVE_ASYNC
    if (has_next) wait_async0();
#else
    if (has_next) {
      int nb = cur ^ 1;
      *(v8h*)&smA[nb][baseA] = a0; *(v8h*)&smA[nb][baseA + 8] = a1;
      *(v8h*)&smB[nb][baseB]      = b0; *(v8h*)&smB[nb][baseB + 8]  = b1;
      *(v8h*)&smB[nb][baseB + 16] = b2; *(v8h*)&smB[nb][baseB + 24] = b3;
    }
#endif
    __syncthreads();
    cur ^= 1;
  }

  const float s = *scaler;
#pragma unroll
  for (int i = 0; i < 4; ++i)
#pragma unroll
    for (int j = 0; j < 4; ++j) {
      float* o = out + (size_t)(m0b + waveM + i * 16 + lhi * 8) * OUT_F
                     + (n0b + waveN + j * 16 + lrow);
#pragma unroll
      for (int r = 0; r < 8; ++r)
        o[(size_t)r * OUT_F] = acc[i][j][r] * s;
    }
}

// ---------------- fallback: fused dequant (ws too small) ----------------
struct RawA { v4f r[4]; };
struct RawB { v4u q[2]; };

__device__ __forceinline__ void stage_A(_Float16* dst, const RawA& ra) {
  v8h h0, h1;
#pragma unroll
  for (int e = 0; e < 4; ++e) {
    h0[e] = (_Float16)ra.r[0][e]; h0[e + 4] = (_Float16)ra.r[1][e];
    h1[e] = (_Float16)ra.r[2][e]; h1[e + 4] = (_Float16)ra.r[3][e];
  }
  *(v8h*)(dst) = h0; *(v8h*)(dst + 8) = h1;
}
__device__ __forceinline__ void stage_B(_Float16* dst, const RawB& rb) {
  *(v8h*)(dst)      = dequant8(rb.q[0][0], rb.q[0][1]);
  *(v8h*)(dst + 8)  = dequant8(rb.q[0][2], rb.q[0][3]);
  *(v8h*)(dst + 16) = dequant8(rb.q[1][0], rb.q[1][1]);
  *(v8h*)(dst + 24) = dequant8(rb.q[1][2], rb.q[1][3]);
}

__global__ __launch_bounds__(256, 1)
void lq_wmma_fused_kernel(const float* __restrict__ x,
                          const int8_t* __restrict__ w,
                          const float* __restrict__ scaler,
                          float* __restrict__ out)
{
  __shared__ _Float16 smA[2][128 * LDA];
  __shared__ _Float16 smB[2][256 * LDB];

  const int t    = threadIdx.x;
  const int lane = t & 31;
  const int wave = t >> 5;
  const int lrow = lane & 15;
  const int lhi  = lane >> 4;

  const int m0b = blockIdx.y * 128;
  const int n0b = blockIdx.x * 256;
  const int waveM = (wave >> 2) * 64;
  const int waveN = (wave & 3) * 64;

  const int rowA  = t >> 1;
  const int halfA = t & 1;
  const int rowB  = t;
  const float*  gA = x + (size_t)(m0b + rowA) * IN_F + halfA * 16;
  const int8_t* gB = w + (size_t)(n0b + rowB) * IN_F;
  _Float16* dA[2] = { &smA[0][rowA * LDA + halfA * 16], &smA[1][rowA * LDA + halfA * 16] };
  _Float16* dB[2] = { &smB[0][rowB * LDB],              &smB[1][rowB * LDB] };

  v8f acc[4][4];
  const v8f vzero = {0.f, 0.f, 0.f, 0.f, 0.f, 0.f, 0.f, 0.f};
#pragma unroll
  for (int i = 0; i < 4; ++i)
#pragma unroll
    for (int j = 0; j < 4; ++j) acc[i][j] = vzero;

  {
    RawA ra;
#pragma unroll
    for (int i = 0; i < 4; ++i) ra.r[i] = *(const v4f*)(gA + 4 * i);
    RawB rb;
    rb.q[0] = ((const v4u*)gB)[0]; rb.q[1] = ((const v4u*)gB)[1];
    stage_A(dA[0], ra); stage_B(dB[0], rb);
  }
  __syncthreads();

  int cur = 0;
  for (int k = 0; k < IN_F; k += KSTEP) {
    const bool has_next = (k + KSTEP < IN_F);
    RawA ra; RawB rb;
    if (has_next) {
#pragma unroll
      for (int i = 0; i < 4; ++i) ra.r[i] = *(const v4f*)(gA + (k + KSTEP) + 4 * i);
      const v4u* p = (const v4u*)(gB + (k + KSTEP));
      rb.q[0] = p[0]; rb.q[1] = p[1];
    }

    v16h a[4], b[4];
#pragma unroll
    for (int i = 0; i < 4; ++i)
      a[i] = frag_from_lds(&smA[cur][(waveM + i * 16 + lrow) * LDA + lhi * 8]);
#pragma unroll
    for (int j = 0; j < 4; ++j)
      b[j] = frag_from_lds(&smB[cur][(waveN + j * 16 + lrow) * LDB + lhi * 8]);

#pragma unroll
    for (int i = 0; i < 4; ++i)
#pragma unroll
      for (int j = 0; j < 4; ++j)
        acc[i][j] = __builtin_amdgcn_wmma_f32_16x16x32_f16(
            false, a[i], false, b[j], (short)0, acc[i][j], false, false);

    if (has_next) { stage_A(dA[cur ^ 1], ra); stage_B(dB[cur ^ 1], rb); }
    __syncthreads();
    cur ^= 1;
  }

  const float s = *scaler;
#pragma unroll
  for (int i = 0; i < 4; ++i)
#pragma unroll
    for (int j = 0; j < 4; ++j) {
      float* o = out + (size_t)(m0b + waveM + i * 16 + lhi * 8) * OUT_F
                     + (n0b + waveN + j * 16 + lrow);
#pragma unroll
      for (int r = 0; r < 8; ++r)
        o[(size_t)r * OUT_F] = acc[i][j][r] * s;
    }
}

extern "C" void kernel_launch(void* const* d_in, const int* in_sizes, int n_in,
                              void* d_out, int out_size, void* d_ws, size_t ws_size,
                              hipStream_t stream) {
  const float*  x      = (const float*)d_in[0];
  const int8_t* w      = (const int8_t*)d_in[1];
  const float*  scaler = (const float*)d_in[2];
  float*        out    = (float*)d_out;

  const int M = in_sizes[0] / IN_F;                    // 8192
  const size_t nW = (size_t)OUT_F * IN_F;              // 45,088,768
  const size_t nX = (size_t)M * IN_F;                  // 33,554,432
  const size_t need = (nW + nX) * sizeof(_Float16);    // ~157 MB

  dim3 grid(OUT_F / 256, M / 128), block(256);

  if (ws_size >= need) {
    _Float16* wsW = (_Float16*)d_ws;
    _Float16* wsX = wsW + nW;
    const int n8w = (int)(nW / 8), n8x = (int)(nX / 8);
    hipLaunchKernelGGL(cvt_w_kernel, dim3((n8w + 255) / 256), block, 0, stream, w, wsW, n8w);
    hipLaunchKernelGGL(cvt_x_kernel, dim3((n8x + 255) / 256), block, 0, stream, x, wsX, n8x);
    hipLaunchKernelGGL(lq_wmma_f16_kernel, grid, block, 0, stream, wsX, wsW, scaler, out);
  } else {
    hipLaunchKernelGGL(lq_wmma_fused_kernel, grid, block, 0, stream, x, w, scaler, out);
  }
}
